// DecoderBlock_90486370992771
// MI455X (gfx1250) — matrix operand order repaired and checked by reading, not compiled, not run on hardware
//
#include <hip/hip_runtime.h>
#include <hip/hip_bf16.h>
#include <math.h>

#define DEV static __device__ __forceinline__

typedef __attribute__((ext_vector_type(16))) __bf16 v16bf;
typedef __attribute__((ext_vector_type(8)))  float  v8f;
typedef __attribute__((ext_vector_type(4)))  unsigned int u32x4;
typedef __attribute__((ext_vector_type(8)))  int i32x8;
typedef __attribute__((ext_vector_type(4)))  int i32x4;

union Frag16 { uint4 u[2]; v16bf v; unsigned short s[16]; };

constexpr int Bc = 8, Sc = 512, Dc = 1024, HIDc = 1024, NHc = 16, Cc = 64;
constexpr int BSc = Bc * Sc;   // 4096 rows

#if __has_builtin(__builtin_amdgcn_tensor_load_to_lds)
#define USE_TDM 1
#else
#define USE_TDM 0
#endif

#if __has_builtin(__builtin_amdgcn_permlane16) && __has_builtin(__builtin_amdgcn_permlanex16)
#define HAS_PERMLANE 1
#else
#define HAS_PERMLANE 0
#endif

DEV unsigned short f2bf(float f) {
  unsigned int u = __float_as_uint(f);
  u += 0x7FFFu + ((u >> 16) & 1u);       // round-to-nearest-even
  return (unsigned short)(u >> 16);
}

// ---- cross-lane butterflies on the VALU (v_permlane16) instead of DS pipe ---
#if HAS_PERMLANE
DEV float plane16(float v, unsigned s0, unsigned s1) {   // within 16-lane rows
  int r = __builtin_amdgcn_permlane16(__float_as_int(v), __float_as_int(v),
                                      (int)s0, (int)s1, false, false);
  return __int_as_float(r);
}
DEV float planex16(float v) {                            // swap 16-lane rows
  int r = __builtin_amdgcn_permlanex16(__float_as_int(v), __float_as_int(v),
                                       (int)0x76543210u, (int)0xfedcba98u,
                                       false, false);
  return __int_as_float(r);
}
DEV float redmax16(float v) {
  v = fmaxf(v, plane16(v, 0x67452301u, 0xefcdab89u));   // xor 1
  v = fmaxf(v, plane16(v, 0x54761032u, 0xdcfe98bau));   // xor 2
  v = fmaxf(v, plane16(v, 0x32107654u, 0xba98fedcu));   // xor 4
  v = fmaxf(v, plane16(v, 0xfedcba98u, 0x76543210u));   // xor 8
  return v;
}
DEV float redsum16(float v) {
  v += plane16(v, 0x67452301u, 0xefcdab89u);
  v += plane16(v, 0x54761032u, 0xdcfe98bau);
  v += plane16(v, 0x32107654u, 0xba98fedcu);
  v += plane16(v, 0xfedcba98u, 0x76543210u);
  return v;
}
DEV float redsum32(float v) { v = redsum16(v); v += planex16(v); return v; }
#else
DEV float redmax16(float v) {
  #pragma unroll
  for (int m = 1; m < 16; m <<= 1) v = fmaxf(v, __shfl_xor(v, m, 32));
  return v;
}
DEV float redsum16(float v) {
  #pragma unroll
  for (int m = 1; m < 16; m <<= 1) v += __shfl_xor(v, m, 32);
  return v;
}
DEV float redsum32(float v) {
  #pragma unroll
  for (int m = 1; m < 32; m <<= 1) v += __shfl_xor(v, m, 32);
  return v;
}
#endif

DEV v8f wmma_bf16(v16bf a, v16bf b, v8f c) {
  return __builtin_amdgcn_wmma_f32_16x16x32_bf16(false, a, false, b, (short)0, c, false, false);
}

#if USE_TDM
// One 2D TDM tile load: tensor bf16 [tdim1][tdim0] row-major (stride0),
// tile [tile1][tile0=64] -> LDS at lds_off; pad 4 DWORDs every 32 DWORDs
// (LDS row stride = 144B, 16B-aligned, 36-bank spread).
DEV void tdm_load_2d(const unsigned short* gtile, unsigned lds_off,
                     unsigned long long tdim0, unsigned long long tdim1,
                     unsigned long long tile0, unsigned long long tile1,
                     unsigned long long stride0) {
  unsigned long long ga = (unsigned long long)gtile;
  u32x4 g0;
  g0.x = 1u;                                     // count=1 (valid descriptor)
  g0.y = lds_off;                                // lds_addr [63:32]
  g0.z = (unsigned)(ga & 0xffffffffu);           // global_addr [95:64]
  g0.w = (unsigned)((ga >> 32) & 0x01ffffffu) | (2u << 30);  // addr hi | type=2
  union { unsigned long long q[4]; i32x8 v; } G1;
  // data_size=1(2B)<<16 | pad_enable<<20 | pad_interval=4(32DW)<<22 | pad_amount=3(4DW)<<25
  unsigned long long flags = (1ull << 16) | (1ull << 20) | (4ull << 22) | (3ull << 25);
  G1.q[0] = flags | ((tdim0 & 0xffffull) << 48);
  G1.q[1] = (tdim0 >> 16) | (tdim1 << 16) | ((tile0 & 0xffffull) << 48);
  G1.q[2] = (tile1 & 0xffffull) | ((stride0 & 0xffffffffull) << 32);
  G1.q[3] = (stride0 >> 32) & 0xffffull;         // tensor_dim1_stride unused (2D)
  i32x4 z4 = {0, 0, 0, 0};
  i32x8 z8 = {0, 0, 0, 0, 0, 0, 0, 0};
  __builtin_amdgcn_tensor_load_to_lds(g0, G1.v, z4, z4, z8, 0);
}
#endif

// ---------------------------------------------------------------- cvt f32->bf16
__global__ void cvt_f32_bf16_kernel(const float* __restrict__ in,
                                    unsigned short* __restrict__ out, int n) {
  int i = blockIdx.x * blockDim.x + threadIdx.x;
  int stride = gridDim.x * blockDim.x;
  for (; i < n; i += stride) out[i] = f2bf(in[i]);
}

// ------------------------------------------- weight transpose + cvt: Wt[n][k]
__global__ __launch_bounds__(256) void transpose_cvt_kernel(
    const float* __restrict__ W, unsigned short* __restrict__ Wt, int K, int N) {
  __shared__ float tile[32][33];
  const int nb = blockIdx.x * 32, kb = blockIdx.y * 32;
  const int tx = threadIdx.x & 31, ty = threadIdx.x >> 5;   // ty 0..7
  #pragma unroll
  for (int r = 0; r < 4; r++)
    tile[ty + r * 8][tx] = W[(size_t)(kb + ty + r * 8) * N + nb + tx];
  __syncthreads();
  #pragma unroll
  for (int r = 0; r < 4; r++)
    Wt[(size_t)(nb + ty + r * 8) * K + kb + tx] = f2bf(tile[tx][ty + r * 8]);
}

// ---------------------------------------------------------------- WMMA GEMM
// Out[M,N] = A_bf16[M,K] @ Wt_bf16[N,K]^T (+bias) (+res); out f32 and/or bf16.
// 256 thr = 8 waves; block tile 64(M) x 128(N); 32x32 per wave; K-stage 64.
// Double-buffered TDM staging: stage i+1 DMAs while stage i computes.
template <bool HB, bool HR, bool OF, bool OH>
__global__ __launch_bounds__(256) void gemm_wmma_kernel(
    const unsigned short* __restrict__ A, const unsigned short* __restrict__ Wt,
    const float* __restrict__ bias, const float* __restrict__ res,
    float* __restrict__ outF, unsigned short* __restrict__ outH,
    int M, int N, int K) {
  __shared__ unsigned short Asb[2][64][72];    // [buf][m][k], 144B rows
  __shared__ unsigned short Bsb[2][128][72];   // [buf][n][k], 144B rows
  const int tid = threadIdx.x;
  const int lane = tid & 31, wid = tid >> 5;
  const int g = lane >> 4, nl = lane & 15;
  const int mw = wid & 1, nw = wid >> 1;
  const int Mb = blockIdx.x * 64, Nb = blockIdx.y * 128;
  const int nst = K / 64;

  v8f zero = {};
  v8f acc[2][2];
  #pragma unroll
  for (int i = 0; i < 2; i++)
    #pragma unroll
    for (int j = 0; j < 2; j++) acc[i][j] = zero;

#if USE_TDM
  if (wid == 0) {     // prologue: DMA stage 0 into buffer 0
    tdm_load_2d(A  + (size_t)Mb * K, (unsigned)(size_t)&Asb[0][0][0], K, M, 64, 64,  K);
    tdm_load_2d(Wt + (size_t)Nb * K, (unsigned)(size_t)&Bsb[0][0][0], K, N, 64, 128, K);
  }
#endif

  for (int i = 0; i < nst; i++) {
    const int bi = i & 1;
    const int kb = i * 64;
#if USE_TDM
    if (wid == 0) {
      if (i + 1 < nst) {      // DMA next stage into the other buffer
        tdm_load_2d(A  + (size_t)Mb * K + kb + 64,
                    (unsigned)(size_t)&Asb[bi ^ 1][0][0], K, M, 64, 64,  K);
        tdm_load_2d(Wt + (size_t)Nb * K + kb + 64,
                    (unsigned)(size_t)&Bsb[bi ^ 1][0][0], K, N, 64, 128, K);
        __builtin_amdgcn_s_wait_tensorcnt(2);   // stage i done, i+1 in flight
      } else {
        __builtin_amdgcn_s_wait_tensorcnt(0);
      }
    }
    __syncthreads();
#else
    #pragma unroll
    for (int j = 0; j < 2; j++) {           // A: 64x64 bf16 = 512 x 16B
      int idx = tid + j * 256;
      int row = idx >> 3, k8 = (idx & 7) * 8;
      *reinterpret_cast<uint4*>(&Asb[bi][row][k8]) =
          *reinterpret_cast<const uint4*>(A + (size_t)(Mb + row) * K + kb + k8);
    }
    #pragma unroll
    for (int j = 0; j < 4; j++) {           // B: 128x64 bf16 = 1024 x 16B
      int idx = tid + j * 256;
      int row = idx >> 3, k8 = (idx & 7) * 8;
      *reinterpret_cast<uint4*>(&Bsb[bi][row][k8]) =
          *reinterpret_cast<const uint4*>(Wt + (size_t)(Nb + row) * K + kb + k8);
    }
    __syncthreads();
#endif
    #pragma unroll
    for (int kk = 0; kk < 64; kk += 32) {
      Frag16 a[2], b[2];
      #pragma unroll
      for (int mi = 0; mi < 2; mi++) {
        int row = mw * 32 + mi * 16 + nl;   // A frag: K = kk + g*8 + {0..7,16..23}
        a[mi].u[0] = *reinterpret_cast<const uint4*>(&Asb[bi][row][kk + g * 8]);
        a[mi].u[1] = *reinterpret_cast<const uint4*>(&Asb[bi][row][kk + g * 8 + 16]);
      }
      #pragma unroll
      for (int ni = 0; ni < 2; ni++) {
        int n = nw * 32 + ni * 16 + nl;     // B frag: K = kk + g*8 + {0..7,16..23}
        b[ni].u[0] = *reinterpret_cast<const uint4*>(&Bsb[bi][n][kk + g * 8]);
        b[ni].u[1] = *reinterpret_cast<const uint4*>(&Bsb[bi][n][kk + g * 8 + 16]);
      }
      #pragma unroll
      for (int mi = 0; mi < 2; mi++)
        #pragma unroll
        for (int ni = 0; ni < 2; ni++)
          acc[mi][ni] = wmma_bf16(a[mi].v, b[ni].v, acc[mi][ni]);
    }
    __syncthreads();
  }

  // straight-line epilogue: incremental addressing, compile-time feature flags
  #pragma unroll
  for (int mi = 0; mi < 2; mi++)
    #pragma unroll
    for (int ni = 0; ni < 2; ni++) {
      int row0 = Mb + mw * 32 + mi * 16 + 8 * g;
      int col  = Nb + nw * 32 + ni * 16 + nl;
      float bv = 0.f;
      if constexpr (HB) bv = bias[col];
      size_t off = (size_t)row0 * N + col;
      #pragma unroll
      for (int r = 0; r < 8; r++, off += (size_t)N) {
        float v = acc[mi][ni][r] + bv;
        if constexpr (HR) v += res[off];
        if constexpr (OF) outF[off] = v;
        if constexpr (OH) outH[off] = f2bf(v);
      }
    }
}

// ---------------------------------------------------------------- self-attn
// Per-token channel attention, causal over C=64, softmax over k<=q.
__global__ __launch_bounds__(256) void self_attn_kernel(
    const float* __restrict__ Q, const float* __restrict__ K,
    const float* __restrict__ V, unsigned short* __restrict__ out) {
  __shared__ float sh[8][3 * 64];
  const int lane = threadIdx.x & 31, wid = threadIdx.x >> 5;
  const int tok = blockIdx.x * 8 + wid;
  float* Qv = sh[wid]; float* Kv = Qv + 64; float* Vv = Kv + 64;
  const size_t base0 = (size_t)tok * HIDc;
  for (int h = 0; h < NHc; h++) {
    const size_t base = base0 + h * Cc;
    Qv[lane]      = Q[base + lane]      * 0.125f;   // 1/sqrt(C)
    Qv[lane + 32] = Q[base + lane + 32] * 0.125f;
    Kv[lane]      = K[base + lane];
    Kv[lane + 32] = K[base + lane + 32];
    Vv[lane]      = V[base + lane];
    Vv[lane + 32] = V[base + lane + 32];
    asm volatile("s_wait_dscnt 0" ::: "memory");
    #pragma unroll
    for (int qq = 0; qq < 2; qq++) {
      int q = lane + qq * 32;
      float qv = Qv[q];
      float m = -1e30f, s = 0.f, acc = 0.f;
      for (int k = 0; k <= q; k++) {                // causal over channels
        float a = qv * Kv[k];
        float mn = fmaxf(m, a);
        float w1 = __expf(m - mn), w2 = __expf(a - mn);
        s = s * w1 + w2;
        acc = acc * w1 + w2 * Vv[k];
        m = mn;
      }
      out[base + q] = f2bf(acc / s);
    }
    asm volatile("s_wait_dscnt 0" ::: "memory");
  }
}

// ---------------------------------------------------------------- cross-attn (flash)
__global__ __launch_bounds__(256) void cross_attn_kernel(
    const unsigned short* __restrict__ Q, const unsigned short* __restrict__ K,
    const unsigned short* __restrict__ V, unsigned short* __restrict__ out) {
  __shared__ unsigned short Ksh[32][72];      // [kv][c]
  __shared__ unsigned short Vsh[64][40];      // transposed [c][kv]
  __shared__ unsigned short Psh[8][16][40];   // per-wave P tile [q][kv]
  const int tid = threadIdx.x;
  const int lane = tid & 31, wid = tid >> 5;
  const int g = lane >> 4, nl = lane & 15;
  const int qt = blockIdx.x, h = blockIdx.y, b = blockIdx.z;
  const int qbase = qt * 128 + wid * 16;
  const size_t headoff = (size_t)h * Cc;

  Frag16 qf[2];                 // Q A-fragments over c=0..31 / 32..63
  {
    const unsigned short* qp =
        Q + (size_t)(b * Sc + qbase + nl) * HIDc + headoff;
    #pragma unroll
    for (int p = 0; p < 2; p++) {
      qf[p].u[0] = *reinterpret_cast<const uint4*>(qp + p * 32 + g * 8);
      qf[p].u[1] = *reinterpret_cast<const uint4*>(qp + p * 32 + g * 8 + 16);
    }
  }
  v8f zero = {};
  v8f o[4];
  #pragma unroll
  for (int t = 0; t < 4; t++) o[t] = zero;
  float mrow[8], lrow[8];
  #pragma unroll
  for (int r = 0; r < 8; r++) { mrow[r] = -1e30f; lrow[r] = 0.f; }
  const float sc = 0.03125f;    // 1/sqrt(HID)

  for (int kt = 0; kt < Sc / 32; kt++) {
    {   // cooperative K/V staging: 32 kv rows x 64 channels
      int kvr = tid >> 3, c8 = (tid & 7) * 8;
      const size_t roff = (size_t)(b * Sc + kt * 32 + kvr) * HIDc + headoff + c8;
      *reinterpret_cast<uint4*>(&Ksh[kvr][c8]) =
          *reinterpret_cast<const uint4*>(K + roff);
      Frag16 vv; vv.u[0] = *reinterpret_cast<const uint4*>(V + roff);
      #pragma unroll
      for (int j = 0; j < 8; j++) Vsh[c8 + j][kvr] = vv.s[j];   // transpose
    }
    __syncthreads();

    v8f s0 = zero, s1 = zero;   // two 16x16 score tiles (kv 0..15, 16..31)
    #pragma unroll
    for (int p = 0; p < 2; p++) {
      Frag16 kf0, kf1;
      kf0.u[0] = *reinterpret_cast<const uint4*>(&Ksh[nl][p * 32 + g * 8]);
      kf0.u[1] = *reinterpret_cast<const uint4*>(&Ksh[nl][p * 32 + g * 8 + 16]);
      kf1.u[0] = *reinterpret_cast<const uint4*>(&Ksh[16 + nl][p * 32 + g * 8]);
      kf1.u[1] = *reinterpret_cast<const uint4*>(&Ksh[16 + nl][p * 32 + g * 8 + 16]);
      s0 = wmma_bf16(qf[p].v, kf0.v, s0);
      s1 = wmma_bf16(qf[p].v, kf1.v, s1);
    }
    // online softmax update; emit P (bf16) into per-wave LDS scratch
    #pragma unroll
    for (int r = 0; r < 8; r++) {
      float a0 = s0[r] * sc, a1 = s1[r] * sc;
      float tmax = redmax16(fmaxf(a0, a1));
      float mn = fmaxf(mrow[r], tmax);
      float corr = __expf(mrow[r] - mn);
      float p0 = __expf(a0 - mn), p1 = __expf(a1 - mn);
      float rs = redsum16(p0 + p1);
      lrow[r] = lrow[r] * corr + rs;
      mrow[r] = mn;
      #pragma unroll
      for (int t = 0; t < 4; t++) o[t][r] *= corr;
      int q = r + 8 * g;
      Psh[wid][q][nl]      = f2bf(p0);
      Psh[wid][q][16 + nl] = f2bf(p1);
    }
    asm volatile("s_wait_dscnt 0" ::: "memory");
    Frag16 pf;                  // P as A-fragment (16 q x 32 kv)
    pf.u[0] = *reinterpret_cast<const uint4*>(&Psh[wid][nl][g * 8]);
    pf.u[1] = *reinterpret_cast<const uint4*>(&Psh[wid][nl][g * 8 + 16]);
    #pragma unroll
    for (int t = 0; t < 4; t++) {
      Frag16 vf;                // V B-fragment: kv contraction, c = t*16+nl
      vf.u[0] = *reinterpret_cast<const uint4*>(&Vsh[t * 16 + nl][g * 8]);
      vf.u[1] = *reinterpret_cast<const uint4*>(&Vsh[t * 16 + nl][g * 8 + 16]);
      o[t] = wmma_bf16(pf.v, vf.v, o[t]);
    }
    __syncthreads();
  }

  #pragma unroll
  for (int r = 0; r < 8; r++) {
    float inv = 1.f / lrow[r];
    int srow = qbase + r + 8 * g;
    unsigned short* op = out + (size_t)(b * Sc + srow) * HIDc + headoff;
    #pragma unroll
    for (int t = 0; t < 4; t++) op[t * 16 + nl] = f2bf(o[t][r] * inv);
  }
}

// ---------------------------------------------------------------- layernorm
template <bool OF, bool OH>
__global__ __launch_bounds__(256) void layernorm_kernel(
    const float* __restrict__ in, const float* __restrict__ gamma,
    const float* __restrict__ beta, float* __restrict__ outF,
    unsigned short* __restrict__ outH) {
  const int lane = threadIdx.x & 31, wid = threadIdx.x >> 5;
  const int row = blockIdx.x * 8 + wid;
  const float* x = in + (size_t)row * Dc;
  float s = 0.f, ss = 0.f;
  #pragma unroll
  for (int i = lane; i < Dc; i += 32) { float v = x[i]; s += v; ss += v * v; }
  s = redsum32(s); ss = redsum32(ss);
  float mu = s * (1.f / Dc);
  float var = ss * (1.f / Dc) - mu * mu;
  float rs = rsqrtf(var + 1e-5f);
  #pragma unroll
  for (int i = lane; i < Dc; i += 32) {
    float v = (x[i] - mu) * rs * gamma[i] + beta[i];
    if constexpr (OF) outF[(size_t)row * Dc + i] = v;
    if constexpr (OH) outH[(size_t)row * Dc + i] = f2bf(v);
  }
}

// ---------------------------------------------------------------- launcher
extern "C" void kernel_launch(void* const* d_in, const int* in_sizes, int n_in,
                              void* d_out, int out_size, void* d_ws, size_t ws_size,
                              hipStream_t stream) {
  const float* x   = (const float*)d_in[0];
  const float* h   = (const float*)d_in[1];
  const float* Wf[10] = { (const float*)d_in[2],  (const float*)d_in[4],
                          (const float*)d_in[6],  (const float*)d_in[8],
                          (const float*)d_in[10], (const float*)d_in[12],
                          (const float*)d_in[14], (const float*)d_in[16],
                          (const float*)d_in[20], (const float*)d_in[22] };
  const float* bq  = (const float*)d_in[3];
  const float* bk  = (const float*)d_in[5];
  const float* bv  = (const float*)d_in[7];
  const float* bo  = (const float*)d_in[9];
  const float* bcq = (const float*)d_in[11];
  const float* bck = (const float*)d_in[13];
  const float* bcv = (const float*)d_in[15];
  const float* bco = (const float*)d_in[17];
  const float* gamma = (const float*)d_in[18];
  const float* beta  = (const float*)d_in[19];
  const float* b1  = (const float*)d_in[21];
  const float* b2  = (const float*)d_in[23];

  char* ws = (char*)d_ws;
  const size_t MBy = 1ull << 20;
  // hand-scheduled aliasing; peak usage 132 MB
  unsigned short* x16  = (unsigned short*)(ws + 0 * MBy);
  unsigned short* h16  = (unsigned short*)(ws + 8 * MBy);
  unsigned short* bufC = (unsigned short*)(ws + 16 * MBy); // att16 -> co16 -> m16
  unsigned short* bufD = (unsigned short*)(ws + 24 * MBy); // y1_16 -> y2_16
  float* Qs = (float*)(ws + 32 * MBy);
  float* Ks = (float*)(ws + 48 * MBy);
  float* Vs = (float*)(ws + 64 * MBy);
  unsigned short* Qc16 = (unsigned short*)(ws + 32 * MBy); // reuse after self-attn
  unsigned short* Kc16 = (unsigned short*)(ws + 40 * MBy);
  unsigned short* Vc16 = (unsigned short*)(ws + 48 * MBy);
  float* t2 = (float*)(ws + 56 * MBy);
  float* t1 = (float*)(ws + 80 * MBy);
  float* y1 = (float*)(ws + 96 * MBy);
  float* y2 = (float*)(ws + 80 * MBy);  // reuse t1
  float* t3 = (float*)(ws + 56 * MBy);  // reuse t2
  unsigned short* Wt16[10];
  for (int i = 0; i < 10; i++)
    Wt16[i] = (unsigned short*)(ws + (112 + 2 * (size_t)i) * MBy);
  float* outp = (float*)d_out;

  dim3 gGemm(BSc / 64, 1024 / 128);
  dim3 gTr(1024 / 32, 1024 / 32);
  dim3 bT(256);
  const int nAct = BSc * Dc;

  // one-time per launch: weights -> bf16 [N,K]; activations -> bf16
  for (int i = 0; i < 10; i++)
    transpose_cvt_kernel<<<gTr, bT, 0, stream>>>(Wf[i], Wt16[i], 1024, 1024);
  cvt_f32_bf16_kernel<<<2048, 256, 0, stream>>>(x, x16, nAct);
  cvt_f32_bf16_kernel<<<2048, 256, 0, stream>>>(h, h16, nAct);

  // self-attention branch
  gemm_wmma_kernel<true, false, true, false><<<gGemm, bT, 0, stream>>>(
      x16, Wt16[0], bq, nullptr, Qs, nullptr, BSc, HIDc, Dc);
  gemm_wmma_kernel<true, false, true, false><<<gGemm, bT, 0, stream>>>(
      x16, Wt16[1], bk, nullptr, Ks, nullptr, BSc, HIDc, Dc);
  gemm_wmma_kernel<true, false, true, false><<<gGemm, bT, 0, stream>>>(
      x16, Wt16[2], bv, nullptr, Vs, nullptr, BSc, HIDc, Dc);
  self_attn_kernel<<<BSc / 8, bT, 0, stream>>>(Qs, Ks, Vs, bufC);
  gemm_wmma_kernel<true, true, true, false><<<gGemm, bT, 0, stream>>>(
      bufC, Wt16[3], bo, x, t1, nullptr, BSc, Dc, HIDc);
  layernorm_kernel<true, true><<<BSc / 8, bT, 0, stream>>>(t1, gamma, beta, y1, bufD);

  // cross-attention branch
  gemm_wmma_kernel<true, false, false, true><<<gGemm, bT, 0, stream>>>(
      bufD, Wt16[4], bcq, nullptr, nullptr, Qc16, BSc, HIDc, Dc);
  gemm_wmma_kernel<true, false, false, true><<<gGemm, bT, 0, stream>>>(
      h16, Wt16[5], bck, nullptr, nullptr, Kc16, BSc, HIDc, Dc);
  gemm_wmma_kernel<true, false, false, true><<<gGemm, bT, 0, stream>>>(
      h16, Wt16[6], bcv, nullptr, nullptr, Vc16, BSc, HIDc, Dc);
  cross_attn_kernel<<<dim3(Sc / 128, NHc, Bc), bT, 0, stream>>>(Qc16, Kc16, Vc16, bufC);
  gemm_wmma_kernel<true, true, true, false><<<gGemm, bT, 0, stream>>>(
      bufC, Wt16[7], bco, y1, t2, nullptr, BSc, Dc, HIDc);
  layernorm_kernel<true, true><<<BSc / 8, bT, 0, stream>>>(t2, gamma, beta, y2, bufD);

  // FFN (two linears, no nonlinearity) + final LN
  gemm_wmma_kernel<true, false, false, true><<<gGemm, bT, 0, stream>>>(
      bufD, Wt16[8], b1, nullptr, nullptr, bufC, BSc, HIDc, Dc);
  gemm_wmma_kernel<true, true, true, false><<<gGemm, bT, 0, stream>>>(
      bufC, Wt16[9], b2, y2, t3, nullptr, BSc, Dc, HIDc);
  layernorm_kernel<true, false><<<BSc / 8, bT, 0, stream>>>(t3, gamma, beta, outp, nullptr);
}
